// AttentionLayers_61100204753003
// MI455X (gfx1250) — matrix operand order repaired and checked by reading, not compiled, not run on hardware
//
#include <hip/hip_runtime.h>
#include <math.h>
#include <stdint.h>

// ---------------------------------------------------------------------------
// AttentionLayers for MI455X (gfx1250, wave32, WMMA + async global->LDS).
//
// Algebraic rewrite (batch-independent attention core):
//   h = LN1(x)
//   Eq[h] = emb @ Wq[h]   (H,F,E)     Ek, Ev likewise           [WMMA bf16]
//   S[h]  = Eq[h] @ Ek[h]^T  (H,F,F)  -- 16 MB, L2-resident     [WMMA bf16]
//   wv[h,g] = sum_e Ev[h,g,e] * Wo[h*E+e]
//   attn_out[b,f] = sum_h softmax_g( (h[b,f]/8)*h[b,g]*S[h,f,g] )
//                          . (h[b,g]*wv[h,g])                   [VALU/exp]
//   x1 = x + attn_out + bo
//   out = x1 + FFN(LN2(x1))                                     [WMMA bf16,
//                                        W tiles staged via async->LDS]
// Scratch required: ~18.5 MB of d_ws.
// ---------------------------------------------------------------------------

#define B_  32
#define F_  1024
#define D_  256
#define H_  4
#define E_  64
#define FH_ 4096

typedef __attribute__((ext_vector_type(16))) __bf16 v16bf;
typedef __attribute__((ext_vector_type(8)))  float  v8f;

__device__ __forceinline__ v8f wmma_bf16(v16bf a, v16bf b, v8f c) {
  return __builtin_amdgcn_wmma_f32_16x16x32_bf16(
      /*neg_a=*/false, a, /*neg_b=*/false, b,
      /*c_mod=*/(short)0, c, /*reuse_a=*/false, /*reuse_b=*/false);
}

// --- CDNA5 async global->LDS copy (16B per lane), tracked by ASYNCcnt ------
__device__ __forceinline__ void async_copy_b128(uint32_t lds_off, const void* gptr) {
  asm volatile("global_load_async_to_lds_b128 %0, %1, off"
               :: "v"(lds_off), "v"((unsigned long long)(uintptr_t)gptr)
               : "memory");
}
__device__ __forceinline__ void wait_async0() {
  asm volatile("s_wait_asynccnt 0" ::: "memory");
}

// --- Fragment loaders (contiguous k order, A and B alike), f32/bf16 -> bf16 inline --------
// A 16x32 (MxK): lanes 0-15 -> M=lane, K = k0+{0..15};
//                lanes 16-31 -> M=lane-16, K = k0+{16..31}.
template <typename T>
__device__ __forceinline__ v16bf ldA(const T* base, int ld, int m0, int k0, int lane) {
  const T* p = base + (size_t)(m0 + (lane & 15)) * ld + k0 + ((lane & 16) ? 16 : 0);
  v16bf a;
#pragma unroll
  for (int i = 0; i < 16; ++i) {
    a[i]     = (__bf16)(float)p[i];
  }
  return a;
}

// B 32x16 (KxN) from row-major [K x N] (ld = N).
template <typename T>
__device__ __forceinline__ v16bf ldB(const T* base, int ld, int k0, int n0, int lane) {
  const T* p = base + (size_t)(k0 + ((lane & 16) ? 16 : 0)) * ld + n0 + (lane & 15);
  v16bf b;
#pragma unroll
  for (int i = 0; i < 16; ++i) b[i] = (__bf16)(float)p[(size_t)i * ld];
  return b;
}

// B 32x16 (KxN) where memory is row-major [N x K] (NT gemm; ld = K stride).
template <typename T>
__device__ __forceinline__ v16bf ldBt(const T* base, int ld, int k0, int n0, int lane) {
  const T* p = base + (size_t)(n0 + (lane & 15)) * ld + k0 + ((lane & 16) ? 16 : 0);
  v16bf b;
#pragma unroll
  for (int i = 0; i < 16; ++i) b[i] = (__bf16)(float)p[i];
  return b;
}

// --- wave32 butterfly reductions (result in all lanes) ---------------------
__device__ __forceinline__ float wave_sum(float v) {
#pragma unroll
  for (int o = 16; o > 0; o >>= 1) v += __shfl_xor(v, o, 32);
  return v;
}
__device__ __forceinline__ float wave_max(float v) {
#pragma unroll
  for (int o = 16; o > 0; o >>= 1) v = fmaxf(v, __shfl_xor(v, o, 32));
  return v;
}
__device__ __forceinline__ float block_sum(float v, float* red) {
  v = wave_sum(v);
  const int tid = threadIdx.x;
  if ((tid & 31) == 0) red[tid >> 5] = v;
  __syncthreads();
  float t = 0.f;
#pragma unroll
  for (int i = 0; i < 8; ++i) t += red[i];
  __syncthreads();
  return t;
}

// --- LayerNorm over rows of (B_, F_); optional bf16 copy -------------------
__global__ void ln_kernel(const float* __restrict__ x, const float* __restrict__ gam,
                          const float* __restrict__ bet, float* __restrict__ out,
                          __bf16* __restrict__ outbf) {
  __shared__ float red[8];
  const int b = blockIdx.x, tid = threadIdx.x;
  const float* row = x + (size_t)b * F_;
  float v[4], s = 0.f, s2 = 0.f;
#pragma unroll
  for (int i = 0; i < 4; ++i) {
    v[i] = row[tid + i * 256];
    s += v[i]; s2 += v[i] * v[i];
  }
  s  = block_sum(s, red);
  s2 = block_sum(s2, red);
  const float mu  = s * (1.f / F_);
  const float var = s2 * (1.f / F_) - mu * mu;
  const float inv = rsqrtf(var + 1e-5f);
#pragma unroll
  for (int i = 0; i < 4; ++i) {
    const int c = tid + i * 256;
    const float yv = (v[i] - mu) * inv * gam[c] + bet[c];
    out[(size_t)b * F_ + c] = yv;
    if (outbf) outbf[(size_t)b * F_ + c] = (__bf16)yv;
  }
}

// --- Eproj[h] = emb(F_,D_) @ W[h](D_,E_) -> bf16 (H,F,E) -------------------
// block 128 (4 waves), wave w owns N-tile w; grid (F_/16, 1, H_)
__global__ void proj_gemm_kernel(const float* __restrict__ emb, const float* __restrict__ W,
                                 __bf16* __restrict__ outp) {
  const int lane = threadIdx.x & 31;
  const int w    = threadIdx.x >> 5;
  const int m0   = blockIdx.x * 16;
  const int n0   = w * 16;
  const int h    = blockIdx.z;
  const float* Wh = W + (size_t)h * D_ * E_;
  v8f acc = {0.f, 0.f, 0.f, 0.f, 0.f, 0.f, 0.f, 0.f};
#pragma unroll
  for (int k0 = 0; k0 < D_; k0 += 32) {
    acc = wmma_bf16(ldA(emb, D_, m0, k0, lane), ldB(Wh, E_, k0, n0, lane), acc);
  }
  __bf16* p = outp + (size_t)h * F_ * E_ +
              (size_t)(m0 + ((lane & 16) ? 8 : 0)) * E_ + n0 + (lane & 15);
#pragma unroll
  for (int r = 0; r < 8; ++r) p[(size_t)r * E_] = (__bf16)acc[r];
}

// --- S[h] = Eq[h] @ Ek[h]^T -> f32 (H,F,F), L2-resident --------------------
// grid (F_/16, F_/64, H_), block 128
__global__ void score_gemm_kernel(const __bf16* __restrict__ Eq, const __bf16* __restrict__ Ek,
                                  float* __restrict__ S) {
  const int lane = threadIdx.x & 31;
  const int w    = threadIdx.x >> 5;
  const int m0   = blockIdx.x * 16;
  const int n0   = (blockIdx.y * 4 + w) * 16;
  const int h    = blockIdx.z;
  const __bf16* A  = Eq + (size_t)h * F_ * E_;
  const __bf16* Bt = Ek + (size_t)h * F_ * E_;
  v8f acc = {0.f, 0.f, 0.f, 0.f, 0.f, 0.f, 0.f, 0.f};
#pragma unroll
  for (int k0 = 0; k0 < E_; k0 += 32)
    acc = wmma_bf16(ldA(A, E_, m0, k0, lane), ldBt(Bt, E_, k0, n0, lane), acc);
  float* p = S + (size_t)h * F_ * F_ +
             (size_t)(m0 + ((lane & 16) ? 8 : 0)) * F_ + n0 + (lane & 15);
#pragma unroll
  for (int r = 0; r < 8; ++r) p[(size_t)r * F_] = acc[r];
}

// --- wv[h,g] = sum_e Ev[h,g,e] * Wo[h*E_+e] --------------------------------
__global__ void wv_kernel(const __bf16* __restrict__ Ev, const float* __restrict__ Wo,
                          float* __restrict__ wv) {
  const int i = blockIdx.x * blockDim.x + threadIdx.x;
  if (i >= H_ * F_) return;
  const int h = i >> 10, g = i & (F_ - 1);
  const __bf16* p = Ev + ((size_t)h * F_ + g) * E_;
  const float*  w = Wo + h * E_;
  float s = 0.f;
#pragma unroll 8
  for (int e = 0; e < E_; ++e) s += (float)p[e] * w[e];
  wv[i] = s;
}

// --- fused softmax + AV + out-proj + residual ------------------------------
// grid (F_), block 256 (8 waves). Block stages S rows + wv table in LDS once
// (S read from L2 exactly once); each wave owns batches b = w, w+8, w+16, w+24
// and uses pure wave-shuffle reductions (no barriers in the hot loop).
__global__ void attn_kernel(const float* __restrict__ x, const float* __restrict__ hbuf,
                            const float* __restrict__ S, const float* __restrict__ wvb,
                            const float* __restrict__ bo, float* __restrict__ x1) {
  __shared__ float Sf[H_][F_];   // 16 KB: S[h][f][:]
  __shared__ float Uv[H_][F_];   // 16 KB: wv[h][:]
  const int f = blockIdx.x, tid = threadIdx.x;
  for (int i = tid; i < H_ * F_; i += 256) {
    const int h = i >> 10, g = i & (F_ - 1);
    Sf[h][g] = S[((size_t)h * F_ + f) * F_ + g];
    Uv[h][g] = wvb[i];
  }
  __syncthreads();
  const int w = tid >> 5, lane = tid & 31;
  for (int b = w; b < B_; b += 8) {
    const float* hrow = hbuf + (size_t)b * F_;
    float hbr[32];
#pragma unroll
    for (int i = 0; i < 32; ++i) hbr[i] = hrow[lane + i * 32];
    const float c = hrow[f] * 0.125f;  // h[b,f] / sqrt(HEAD_SIZE)
    float acc = 0.f;
    for (int h = 0; h < H_; ++h) {
      float lmax = -3.402823466e38f;
#pragma unroll
      for (int i = 0; i < 32; ++i)
        lmax = fmaxf(lmax, c * hbr[i] * Sf[h][lane + i * 32]);
      const float m = wave_max(lmax);
      float ls = 0.f, ln = 0.f;
#pragma unroll
      for (int i = 0; i < 32; ++i) {
        const int g = lane + i * 32;
        const float p = __expf(c * hbr[i] * Sf[h][g] - m);
        ls += p;
        ln += p * hbr[i] * Uv[h][g];
      }
      ls = wave_sum(ls);
      ln = wave_sum(ln);
      acc += ln / ls;
    }
    if (lane == 0) x1[(size_t)b * F_ + f] = x[(size_t)b * F_ + f] + acc + bo[0];
  }
}

// --- FFN GEMM1: y = gelu(h2 @ W1 + bf1), bf16 (B_, FH_) --------------------
// grid (FH_/64, B_/16), block 128. W1 32x64 f32 tile staged via async->LDS.
__global__ void ffn1_kernel(const __bf16* __restrict__ h2, const float* __restrict__ W1,
                            const float* __restrict__ bf1, __bf16* __restrict__ y) {
  __shared__ float tileB[32][64];                 // 8 KB staging tile
  const int tid  = threadIdx.x;
  const int lane = tid & 31;
  const int w    = tid >> 5;
  const int nbase = blockIdx.x * 64;
  const int m0    = blockIdx.y * 16;
  const uint32_t lbase = (uint32_t)(uintptr_t)(&tileB[0][0]);
  const int rr = tid >> 2;                        // tile row 0..31
  const int qq = tid & 3;                         // 64B quarter of a 256B row
  v8f acc = {0.f, 0.f, 0.f, 0.f, 0.f, 0.f, 0.f, 0.f};
  for (int k0 = 0; k0 < F_; k0 += 32) {
    __syncthreads();                              // prior reads done
    const float*  gsrc = W1 + (size_t)(k0 + rr) * FH_ + nbase + qq * 16;
    const uint32_t ldst = lbase + (uint32_t)(rr * 64 + qq * 16) * 4u;
    async_copy_b128(ldst +  0, gsrc +  0);
    async_copy_b128(ldst + 16, gsrc +  4);
    async_copy_b128(ldst + 32, gsrc +  8);
    async_copy_b128(ldst + 48, gsrc + 12);
    wait_async0();
    __syncthreads();                              // tile visible to all waves
    v16bf bf;
    const int nl = w * 16 + (lane & 15);
    const int kb = (lane & 16) ? 16 : 0;
#pragma unroll
    for (int i = 0; i < 16; ++i) bf[i] = (__bf16)tileB[kb + i][nl];
    acc = wmma_bf16(ldA(h2, F_, m0, k0, lane), bf, acc);
  }
  const int n  = nbase + w * 16 + (lane & 15);
  const int mo = (lane & 16) ? 8 : 0;
#pragma unroll
  for (int r = 0; r < 8; ++r) {
    float t = acc[r] + bf1[n];
    t = 0.5f * t * (1.0f + erff(t * 0.70710678118654752f));  // exact gelu
    y[(size_t)(m0 + mo + r) * FH_ + n] = (__bf16)t;
  }
}

// --- FFN GEMM2 + residual: out = x1 + y @ W2 + bf2 -------------------------
// grid (F_/64, B_/16), block 128. Same async->LDS staging for W2.
__global__ void ffn2_kernel(const __bf16* __restrict__ y, const float* __restrict__ W2,
                            const float* __restrict__ bf2, const float* __restrict__ x1,
                            float* __restrict__ out) {
  __shared__ float tileB[32][64];
  const int tid  = threadIdx.x;
  const int lane = tid & 31;
  const int w    = tid >> 5;
  const int nbase = blockIdx.x * 64;
  const int m0    = blockIdx.y * 16;
  const uint32_t lbase = (uint32_t)(uintptr_t)(&tileB[0][0]);
  const int rr = tid >> 2;
  const int qq = tid & 3;
  v8f acc = {0.f, 0.f, 0.f, 0.f, 0.f, 0.f, 0.f, 0.f};
  for (int k0 = 0; k0 < FH_; k0 += 32) {
    __syncthreads();
    const float*  gsrc = W2 + (size_t)(k0 + rr) * F_ + nbase + qq * 16;
    const uint32_t ldst = lbase + (uint32_t)(rr * 64 + qq * 16) * 4u;
    async_copy_b128(ldst +  0, gsrc +  0);
    async_copy_b128(ldst + 16, gsrc +  4);
    async_copy_b128(ldst + 32, gsrc +  8);
    async_copy_b128(ldst + 48, gsrc + 12);
    wait_async0();
    __syncthreads();
    v16bf bf;
    const int nl = w * 16 + (lane & 15);
    const int kb = (lane & 16) ? 16 : 0;
#pragma unroll
    for (int i = 0; i < 16; ++i) bf[i] = (__bf16)tileB[kb + i][nl];
    acc = wmma_bf16(ldA(y, FH_, m0, k0, lane), bf, acc);
  }
  const int n  = nbase + w * 16 + (lane & 15);
  const int mo = (lane & 16) ? 8 : 0;
#pragma unroll
  for (int r = 0; r < 8; ++r) {
    const int row = m0 + mo + r;
    out[(size_t)row * F_ + n] = x1[(size_t)row * F_ + n] + acc[r] + bf2[n];
  }
}

// ---------------------------------------------------------------------------
extern "C" void kernel_launch(void* const* d_in, const int* in_sizes, int n_in,
                              void* d_out, int out_size, void* d_ws, size_t ws_size,
                              hipStream_t stream) {
  const float* x   = (const float*)d_in[0];
  const float* emb = (const float*)d_in[1];
  const float* Wq  = (const float*)d_in[2];
  const float* Wk  = (const float*)d_in[3];
  const float* Wv  = (const float*)d_in[4];
  const float* Wo  = (const float*)d_in[5];
  const float* bo  = (const float*)d_in[6];
  const float* g1  = (const float*)d_in[7];
  const float* b1  = (const float*)d_in[8];
  const float* g2  = (const float*)d_in[9];
  const float* b2  = (const float*)d_in[10];
  const float* W1  = (const float*)d_in[11];
  const float* bf1 = (const float*)d_in[12];
  const float* W2  = (const float*)d_in[13];
  const float* bf2 = (const float*)d_in[14];
  float* out = (float*)d_out;
  (void)in_sizes; (void)n_in; (void)out_size; (void)ws_size;

  char* wsb = (char*)d_ws;
  size_t off = 0;
  auto alloc = [&](size_t bytes) -> void* {
    void* p = wsb + off;
    off += (bytes + 255) & ~(size_t)255;
    return p;
  };
  float*  hbuf = (float*) alloc((size_t)B_ * F_ * 4);
  __bf16* Eq   = (__bf16*)alloc((size_t)H_ * F_ * E_ * 2);
  __bf16* Ek   = (__bf16*)alloc((size_t)H_ * F_ * E_ * 2);
  __bf16* Ev   = (__bf16*)alloc((size_t)H_ * F_ * E_ * 2);
  float*  wvb  = (float*) alloc((size_t)H_ * F_ * 4);
  float*  S    = (float*) alloc((size_t)H_ * F_ * F_ * 4);  // 16 MB, L2-resident
  float*  x1   = (float*) alloc((size_t)B_ * F_ * 4);
  float*  h2f  = (float*) alloc((size_t)B_ * F_ * 4);
  __bf16* h2b  = (__bf16*)alloc((size_t)B_ * F_ * 2);
  __bf16* y    = (__bf16*)alloc((size_t)B_ * FH_ * 2);

  ln_kernel<<<B_, 256, 0, stream>>>(x, g1, b1, hbuf, (__bf16*)nullptr);
  proj_gemm_kernel<<<dim3(F_ / 16, 1, H_), 128, 0, stream>>>(emb, Wq, Eq);
  proj_gemm_kernel<<<dim3(F_ / 16, 1, H_), 128, 0, stream>>>(emb, Wk, Ek);
  proj_gemm_kernel<<<dim3(F_ / 16, 1, H_), 128, 0, stream>>>(emb, Wv, Ev);
  wv_kernel<<<(H_ * F_ + 255) / 256, 256, 0, stream>>>(Ev, Wo, wvb);
  score_gemm_kernel<<<dim3(F_ / 16, F_ / 64, H_), 128, 0, stream>>>(Eq, Ek, S);
  attn_kernel<<<F_, 256, 0, stream>>>(x, hbuf, S, wvb, bo, x1);
  ln_kernel<<<B_, 256, 0, stream>>>(x1, g2, b2, h2f, h2b);
  ffn1_kernel<<<dim3(FH_ / 64, B_ / 16), 128, 0, stream>>>(h2b, W1, bf1, y);
  ffn2_kernel<<<dim3(F_ / 64, B_ / 16), 128, 0, stream>>>(y, W2, bf2, x1, out);
}
